// BertUnpadSelfAttention_41592463294984
// MI455X (gfx1250) — compile-verified, matrix operand order checked by reading, not run on hardware
//
#include <hip/hip_runtime.h>
#include <hip/hip_bf16.h>
#include <math.h>

// ---------------- CDNA5 WMMA types/helpers ----------------
typedef __attribute__((ext_vector_type(16))) _Float16 v16h;
typedef __attribute__((ext_vector_type(8)))  _Float16 v8h;
typedef __attribute__((ext_vector_type(8)))  float    v8f;

__device__ __forceinline__ v8f wmma16x16x32(v16h a, v16h b, v8f c) {
    // D = A(16x32) * B(32x16) + C, f16 in / f32 accumulate
    return __builtin_amdgcn_wmma_f32_16x16x32_f16(
        /*neg_a=*/false, a, /*neg_b=*/false, b,
        /*c_mod=*/(short)0, c, /*reuse_a=*/false, /*reuse_b=*/false);
}

// Load one f16 A/B fragment from a 32-half contiguous k-chunk.
// ISA layout (16-bit 16x32 A): lane half hf: elems 0..7 <- k = 8*hf+0..7,
// elems 8..15 <- k = 16+8*hf+0..7.  B(32x16) mirrors with lane = column.
__device__ __forceinline__ v16h load_frag_h(const _Float16* __restrict__ p, int hf) {
    const v8h lo = *(const v8h*)(p + 8 * hf);        // global_load_b128
    const v8h hi = *(const v8h*)(p + 16 + 8 * hf);   // global_load_b128
    v16h r;
#pragma unroll
    for (int i = 0; i < 8; ++i) {
        r[i]     = lo[i];
        r[8 + i] = hi[i];
    }
    return r;
}

#define HID   768
#define QKVW  2304          // 3*HID
#define QKW   1536          // Q|K row stride in scratch
#define NHEAD 12
#define HD    64
#define SMAX  1024

// ---------------- Kernel 0: f32 -> f16 conversion (one-time) ----------------
__global__ void __launch_bounds__(256)
cvt_f32_to_f16_kernel(const float* __restrict__ src, _Float16* __restrict__ dst, int n) {
    int i8 = (blockIdx.x * blockDim.x + threadIdx.x) * 8;
    for (; i8 + 8 <= n; i8 += gridDim.x * blockDim.x * 8) {
        const float4 a = *(const float4*)(src + i8);
        const float4 b = *(const float4*)(src + i8 + 4);
        v8h o;
        o[0] = (_Float16)a.x; o[1] = (_Float16)a.y;
        o[2] = (_Float16)a.z; o[3] = (_Float16)a.w;
        o[4] = (_Float16)b.x; o[5] = (_Float16)b.y;
        o[6] = (_Float16)b.z; o[7] = (_Float16)b.w;
        *(v8h*)(dst + i8) = o;
    }
}

// ---------------- Kernel 1: QKV projection (f16 operands, f32 accum) --------
// One wave computes a 16x64 tile, explicit ping-pong double buffering over
// 32-wide k-chunks (step-64 loop, no cross-iteration register rotation).
__global__ void __launch_bounds__(32)
qkv_gemm_kernel(const _Float16* __restrict__ X, const _Float16* __restrict__ W,
                const float* __restrict__ Wb, _Float16* __restrict__ qk,
                _Float16* __restrict__ Vt, int nnz, int nnzp) {
    const int lane = threadIdx.x;
    const int lid  = lane & 15;
    const int hf   = lane >> 4;

    const int m0 = blockIdx.x * 16;
    const int j0 = blockIdx.y * 64;

    int am = m0 + lid;                          // A row (clamped for tail tiles)
    if (am >= nnz) am = nnz - 1;
    const _Float16* arow = X + (size_t)am * HID;
    const _Float16* b0r  = W + (size_t)(j0 + lid) * HID;   // B columns j0+lid+16*nt
    const _Float16* b1r  = b0r + (size_t)16 * HID;
    const _Float16* b2r  = b0r + (size_t)32 * HID;
    const _Float16* b3r  = b0r + (size_t)48 * HID;

    v8f acc0 = {}, acc1 = {}, acc2 = {}, acc3 = {};

    // buffer A preload (k = 0)
    v16h a0  = load_frag_h(arow, hf);
    v16h b00 = load_frag_h(b0r, hf);
    v16h b01 = load_frag_h(b1r, hf);
    v16h b02 = load_frag_h(b2r, hf);
    v16h b03 = load_frag_h(b3r, hf);

    for (int k0 = 0; k0 < HID; k0 += 64) {
        // load buffer B (k0+32, always < HID since HID is a multiple of 64)
        const int ka = k0 + 32;
        v16h a1  = load_frag_h(arow + ka, hf);
        v16h b10 = load_frag_h(b0r + ka, hf);
        v16h b11 = load_frag_h(b1r + ka, hf);
        v16h b12 = load_frag_h(b2r + ka, hf);
        v16h b13 = load_frag_h(b3r + ka, hf);

        acc0 = wmma16x16x32(a0, b00, acc0);
        acc1 = wmma16x16x32(a0, b01, acc1);
        acc2 = wmma16x16x32(a0, b02, acc2);
        acc3 = wmma16x16x32(a0, b03, acc3);

        // reload buffer A (k0+64; branchless wrap to 0 on the last chunk)
        const int kb = (k0 + 64 < HID) ? (k0 + 64) : 0;
        a0  = load_frag_h(arow + kb, hf);
        b00 = load_frag_h(b0r + kb, hf);
        b01 = load_frag_h(b1r + kb, hf);
        b02 = load_frag_h(b2r + kb, hf);
        b03 = load_frag_h(b3r + kb, hf);

        acc0 = wmma16x16x32(a1, b10, acc0);
        acc1 = wmma16x16x32(a1, b11, acc1);
        acc2 = wmma16x16x32(a1, b12, acc2);
        acc3 = wmma16x16x32(a1, b13, acc3);
    }

    if (j0 < QKW) {
        // Q,K columns: row-major scatter (lanes 0-15 cover 32B runs per row)
#pragma unroll
        for (int nt = 0; nt < 4; ++nt) {
            const v8f acc = (nt == 0) ? acc0 : (nt == 1) ? acc1 : (nt == 2) ? acc2 : acc3;
            const int j = j0 + 16 * nt + lid;
            const float bj = Wb[j];
#pragma unroll
            for (int r = 0; r < 8; ++r) {
                const int m = m0 + r + 8 * hf;   // C layout: lane=N(col), elem r -> M=r+8*half
                if (m < nnz)
                    qk[(size_t)m * QKW + j] = (_Float16)(acc[r] + bj);
            }
        }
    } else {
        // V columns: transposed store, 8 consecutive tokens per lane -> 16B store
#pragma unroll
        for (int nt = 0; nt < 4; ++nt) {
            const v8f acc = (nt == 0) ? acc0 : (nt == 1) ? acc1 : (nt == 2) ? acc2 : acc3;
            const int j = j0 + 16 * nt + lid;
            const float bj = Wb[j];
            v8h o;
#pragma unroll
            for (int r = 0; r < 8; ++r) o[r] = (_Float16)(acc[r] + bj);
            *(v8h*)(Vt + (size_t)(j - QKW) * nnzp + m0 + 8 * hf) = o;
        }
    }
}

// ---------------- Kernel 2: flash attention on unpadded tokens ----------------
// One wave per (batch b, head h, 16-query tile). Computes S^T = K*Q^T so that
// the probability tile's C-layout coincides with the A-fragment layout for P*V.
// K fragments are pipelined one 32-key block ahead (ping-pong KSet).
struct KSet { v16h f00, f01, f10, f11; };

__device__ __forceinline__ void load_kset(KSet& ks, int t0, int L, int lid, int hf,
                                          const _Float16* __restrict__ kbase) {
    int ta = t0 + lid;      if (ta >= L) ta = L - 1;
    int tb = t0 + 16 + lid; if (tb >= L) tb = L - 1;
    const _Float16* pa = kbase + (size_t)ta * QKW;
    const _Float16* pb = kbase + (size_t)tb * QKW;
    ks.f00 = load_frag_h(pa, hf);
    ks.f01 = load_frag_h(pa + 32, hf);
    ks.f10 = load_frag_h(pb, hf);
    ks.f11 = load_frag_h(pb + 32, hf);
}

__device__ __forceinline__ void attn_block(
    int t0, int next_t0, int L, int lid, int hf,
    const float* __restrict__ bias_b, const _Float16* __restrict__ kbase,
    const _Float16* __restrict__ vtbase, int nnzp,
    const v16h& bq0, const v16h& bq1, const KSet& kc, KSet& kn,
    float& m_s, float& l_s, v8f& o0, v8f& o1, v8f& o2, v8f& o3) {
    const float SCALE = 0.125f;                 // 1/sqrt(64)

    // V fragments issued early: B[k=key][n=d], token-major => b128 loads
    v16h bV[4];
#pragma unroll
    for (int c = 0; c < 4; ++c) {
        const _Float16* vc = vtbase + (size_t)(16 * c + lid) * nnzp + t0;
        bV[c] = load_frag_h(vc, hf);
        __builtin_prefetch(vc + 32, 0, 1);      // next block's V (global_prefetch_b8)
    }

    // scores with the pre-loaded K set
    v8f sA = {}, sB = {};
    sA = wmma16x16x32(kc.f00, bq0, sA);
    sA = wmma16x16x32(kc.f01, bq1, sA);
    sB = wmma16x16x32(kc.f10, bq0, sB);
    sB = wmma16x16x32(kc.f11, bq1, sB);

    // pipeline: issue next block's K loads now (kc is dead past this point)
    load_kset(kn, next_t0, L, lid, hf, kbase);

    // bias for key t0+lane lives in this lane; broadcast per element below
    float bv = (t0 + lid + 16 * hf < L) ? bias_b[t0 + lid + 16 * hf] : 0.0f;

    // scale + bias + length mask.  S^T C-layout: elem r -> key t0 + r + 8*hf,
    // lane -> query s0 + lid.
    float sc0[8], sc1[8];
#pragma unroll
    for (int r = 0; r < 8; ++r) {
        const int k0i = t0 + r + 8 * hf;
        const int k1i = k0i + 16;
        const float b0 = __shfl(bv, r + 8 * hf, 32);
        const float b1 = __shfl(bv, 16 + r + 8 * hf, 32);
        sc0[r] = (k0i < L) ? (sA[r] * SCALE + b0) : -1e30f;
        sc1[r] = (k1i < L) ? (sB[r] * SCALE + b1) : -1e30f;
    }

    // online softmax: per-query = per-lane reduce (16 elems) + partner half
    float mx = -3e38f;
#pragma unroll
    for (int r = 0; r < 8; ++r) mx = fmaxf(mx, fmaxf(sc0[r], sc1[r]));
    mx = fmaxf(mx, __shfl_xor(mx, 16, 32));
    const float m_new = fmaxf(m_s, mx);
    const float corr  = __expf(m_s - m_new);

    float rs = 0.0f;
    float p0[8], p1[8];
#pragma unroll
    for (int r = 0; r < 8; ++r) {
        p0[r] = __expf(sc0[r] - m_new);
        p1[r] = __expf(sc1[r] - m_new);
        rs += p0[r] + p1[r];
    }
    rs += __shfl_xor(rs, 16, 32);
    l_s = l_s * corr + rs;
    m_s = m_new;

    // rescale O: elem r of O corresponds to query r + 8*hf -> fetch its corr
    float crr[8];
#pragma unroll
    for (int r = 0; r < 8; ++r) crr[r] = __shfl(corr, r + 8 * hf, 32);
#pragma unroll
    for (int r = 0; r < 8; ++r) {
        o0[r] *= crr[r]; o1[r] *= crr[r]; o2[r] *= crr[r]; o3[r] *= crr[r];
    }

    // P as A fragment: lane = query row, elems 0..7 = keys 8*hf+0..7 (== p0),
    // elems 8..15 = keys 16+8*hf+0..7 (== p1).  Exact layout match, no shuffle.
    v16h aP;
#pragma unroll
    for (int r = 0; r < 8; ++r) {
        aP[r]     = (_Float16)p0[r];
        aP[8 + r] = (_Float16)p1[r];
    }

    o0 = wmma16x16x32(aP, bV[0], o0);
    o1 = wmma16x16x32(aP, bV[1], o1);
    o2 = wmma16x16x32(aP, bV[2], o2);
    o3 = wmma16x16x32(aP, bV[3], o3);
}

__global__ void __launch_bounds__(32)
attn_kernel(const _Float16* __restrict__ qk, const _Float16* __restrict__ Vt,
            const float* __restrict__ bias, const int* __restrict__ cu,
            float* __restrict__ out, int nnzp) {
    const int b  = blockIdx.z;
    const int h  = blockIdx.y;
    const int qt = blockIdx.x;

    const int r0 = cu[b];
    const int L  = cu[b + 1] - r0;
    const int s0 = qt * 16;
    if (s0 >= L) return;                       // wave-uniform exit: EXEC stays all-1

    const int lane = threadIdx.x;
    const int lid  = lane & 15;
    const int hf   = lane >> 4;

    const float* bias_b = bias + (size_t)b * SMAX;

    // Q fragments (B operand of S^T = K * Q^T): column = query s0+lid
    int qrow = s0 + lid; if (qrow >= L) qrow = L - 1;
    const _Float16* qp = qk + (size_t)(r0 + qrow) * QKW + h * HD;
    const v16h bq0 = load_frag_h(qp, hf);           // d = 0..31
    const v16h bq1 = load_frag_h(qp + 32, hf);      // d = 32..63

    float m_s = -3e38f, l_s = 0.0f;
    v8f o0 = {}, o1 = {}, o2 = {}, o3 = {};

    const _Float16* kbase  = qk + (size_t)r0 * QKW + HID + h * HD;          // K rows
    const _Float16* vtbase = Vt + (size_t)(h * HD) * nnzp + r0;             // V columns

    // preload first K block, then pair-process 64 keys per iteration.
    // A trailing padded 32-key block is processed branchlessly: all its
    // probabilities are exactly 0 (exp underflow of -1e30 masks), loads are
    // clamped (K) or stay inside the Vt halo (V).
    KSet kA, kB;
    load_kset(kA, 0, L, lid, hf, kbase);

    for (int t0 = 0; t0 < L; t0 += 64) {
        attn_block(t0,      t0 + 32, L, lid, hf, bias_b, kbase, vtbase, nnzp,
                   bq0, bq1, kA, kB, m_s, l_s, o0, o1, o2, o3);
        attn_block(t0 + 32, t0 + 64, L, lid, hf, bias_b, kbase, vtbase, nnzp,
                   bq0, bq1, kB, kA, m_s, l_s, o0, o1, o2, o3);
    }

    // epilogue: O C-layout lane = d-col (within 16-chunk), elem r -> query r+8*hf
    const float linv = 1.0f / l_s;
    float lrr[8];
#pragma unroll
    for (int r = 0; r < 8; ++r) lrr[r] = __shfl(linv, r + 8 * hf, 32);

#pragma unroll
    for (int r = 0; r < 8; ++r) {
        const int q = s0 + r + 8 * hf;
        if (q < L) {
            float* op = out + (size_t)(r0 + q) * HID + h * HD + lid;
            op[0]  = o0[r] * lrr[r];
            op[16] = o1[r] * lrr[r];
            op[32] = o2[r] * lrr[r];
            op[48] = o3[r] * lrr[r];
        }
    }
}

// ---------------- launcher ----------------
extern "C" void kernel_launch(void* const* d_in, const int* in_sizes, int n_in,
                              void* d_out, int out_size, void* d_ws, size_t ws_size,
                              hipStream_t stream) {
    const float* X    = (const float*)d_in[0];   // hidden_states [nnz, 768]
    const float* W    = (const float*)d_in[1];   // Wqkv_w [2304, 768]
    const float* Wb   = (const float*)d_in[2];   // Wqkv_b [2304]
    const int*   cu   = (const int*)  d_in[3];   // cu_seqlens [B+1]
    const float* bias = (const float*)d_in[7];   // bias [B,1,1,S]

    const int nnz   = in_sizes[0] / HID;
    const int mtile = (nnz + 15) / 16;
    const int nnzp  = mtile * 16;                // padded token count (Vt stride)
    const int nX    = nnz * HID;
    const int nW    = QKVW * HID;

    // workspace partition (all 16B-aligned halves): Xh | Wh | qk | Vt(+64 halo)
    _Float16* Xh = (_Float16*)d_ws;
    _Float16* Wh = Xh + nX;
    _Float16* qk = Wh + nW;                      // [nnz, 1536] row-major Q|K
    _Float16* Vt = qk + (size_t)nnz * QKW;       // [768, nnzp] token-major V

    cvt_f32_to_f16_kernel<<<(nX / 8 + 255) / 256, 256, 0, stream>>>(X, Xh, nX);
    cvt_f32_to_f16_kernel<<<(nW / 8 + 255) / 256, 256, 0, stream>>>(W, Wh, nW);

    dim3 g1(mtile, QKVW / 64);                   // (Mtiles, 36)
    qkv_gemm_kernel<<<g1, 32, 0, stream>>>(Xh, Wh, Wb, qk, Vt, nnz, nnzp);

    dim3 g2(SMAX / 16, NHEAD, 4);                // (64 qtiles max, 12 heads, B)
    attn_kernel<<<g2, 32, 0, stream>>>(qk, Vt, bias, cu, (float*)d_out, nnzp);
}
